// QuantizedD4Linear_8512625180849
// MI455X (gfx1250) — matrix operand order repaired and checked by reading, not compiled, not run on hardware
//
#include <hip/hip_runtime.h>
#include <hip/hip_bf16.h>

// ---------------------------------------------------------------------------
// QuIP#-style D4 quantized linear:
//   out = SV * FWHT( (FWHT(x*SU)/1024 @ W^T) * Wscale*1024 )
// W decoded on the fly from 8-bit codes via a 256x4 fp16 LDS codebook.
// GEMM M=8192, N=4096, K=4096 on v_wmma_f32_16x16x32_f16.
// A-tiles staged into LDS by the Tensor Data Mover (tensor_load_to_lds),
// double-buffered, padded by the TDM to a conflict-free 40-half stride.
// ---------------------------------------------------------------------------

typedef _Float16     v16h __attribute__((ext_vector_type(16)));
typedef float        v8f  __attribute__((ext_vector_type(8)));
typedef unsigned int v4u  __attribute__((ext_vector_type(4)));
typedef int          v8i  __attribute__((ext_vector_type(8)));
typedef int          v4i  __attribute__((ext_vector_type(4)));

union Frag16 { v16h h; uint4 q[2]; };

#define D_IN   4096      // K
#define M_OUT  4096      // N (rows of W)
#define NROWS  8192      // M (4 * 2048)
#define TM     128
#define TN     256
#define TK     32
#define LDT    40        // LDS tile stride in halves: 80B -> 16B aligned, conflict-free

// ---------------------------------------------------------------------------
// Pass A: per-row  x*SU -> FWHT(4096) -> *(1/64)*(1/1024) -> fp16
// ---------------------------------------------------------------------------
__global__ __launch_bounds__(256) void qd4_fwht_in(
    const float* __restrict__ in, const float* __restrict__ SU,
    _Float16* __restrict__ A)
{
    __shared__ float s[D_IN];
    const int row = blockIdx.x;
    const size_t base = (size_t)row * D_IN;
    for (int i = threadIdx.x; i < D_IN; i += 256)
        s[i] = in[base + i] * SU[i];
    __syncthreads();
    for (int h = 1; h < D_IN; h <<= 1) {
        for (int p = threadIdx.x; p < D_IN / 2; p += 256) {
            const int i = ((p & ~(h - 1)) << 1) | (p & (h - 1));
            const int j = i + h;
            const float a = s[i], b = s[j];
            s[i] = a + b;
            s[j] = a - b;
        }
        __syncthreads();
    }
    const float sc = (1.0f / 64.0f) * (1.0f / 1024.0f); // 1/sqrt(4096) * 1/_NUM_SCALE
    for (int i = threadIdx.x; i < D_IN; i += 256)
        A[base + i] = (_Float16)(s[i] * sc);
}

// ---------------------------------------------------------------------------
// TDM: load a TM x TK fp16 tile (row-major, row stride D_IN) into LDS,
// padding 4 DWORDs after every 16 DWORDs so each LDS row occupies LDT=40
// halves (conflict-free for the per-lane ds_load_b128 fragment reads).
// D# per CDNA5 ISA 8.3/8.4; 6-arg clang builtin (g0, g1, g2, g3, g4, cpol).
// ---------------------------------------------------------------------------
__device__ __forceinline__ void tdm_load_a_tile(const _Float16* gsrc, void* lds_dst)
{
    const unsigned long long ga = (unsigned long long)(uintptr_t)gsrc;
    v4u g0;
    g0.x = 1u;                                        // count=1, user descriptor
    g0.y = (unsigned int)(uintptr_t)lds_dst;          // lds_addr = addr[31:0]
    g0.z = (unsigned int)ga;                          // global_addr[31:0]
    g0.w = (unsigned int)((ga >> 32) & 0x01ffffffu)   // global_addr[56:32]
         | (2u << 30);                                // type = 2 ("image")
    v8i g1;
    g1[0] = (1 << 16)                 // data_size = 1 -> 2-byte elements
          | (1 << 20)                 // pad_enable
          | (3 << 22)                 // pad_interval: every 16 DWORDs (64B row)
          | (3 << 25);                // pad_amount: 4 DWORDs (16B -> stride 80B)
    g1[1] = (D_IN & 0xffff) << 16;                            // tensor_dim0[15:0]
    g1[2] = ((D_IN >> 16) & 0xffff) | ((NROWS & 0xffff) << 16); // dim0 hi | dim1 lo
    g1[3] = ((NROWS >> 16) & 0xffff) | (TK << 16);            // dim1 hi | tile_dim0
    g1[4] = TM;                                               // tile_dim1 (tile_dim2=0)
    g1[5] = D_IN;                                             // tensor_dim0_stride lo
    g1[6] = 0;                                                // stride hi | dim1_stride lo
    g1[7] = 0;
    const v4i gz4 = {0, 0, 0, 0};                             // 2D: groups 2/3 unused
    const v8i gz8 = {0, 0, 0, 0, 0, 0, 0, 0};
    __builtin_amdgcn_tensor_load_to_lds(g0, g1, gz4, gz4, gz8, 0);
}

// ---------------------------------------------------------------------------
// Pass B: Z[m][n] = sum_k A[m][k] * W[n][k], W decoded from Qidxs + codebook.
// Block tile 128x256, 8 wave32 (2 in M x 4 in N), wave tile 64x64 = 4x4 frags
// -> 16 v_wmma per wave per K-step against 16 ds_load_b128 fragment loads.
// ---------------------------------------------------------------------------
__global__ __launch_bounds__(256) void qd4_gemm(
    const _Float16* __restrict__ A,      // [NROWS][D_IN] fp16
    const int*      __restrict__ Qidxs,  // [M_OUT][D_IN/4] codes
    const _Float16* __restrict__ CB,     // [256][4] fp16 codebook
    _Float16*       __restrict__ Z)      // [NROWS][M_OUT] fp16
{
    __shared__ _Float16 Atile[2][TM * LDT];   // double-buffered TDM destination
    __shared__ _Float16 Btile[TN * LDT];
    __shared__ uint2    cb[256];

    const int t = threadIdx.x;
    cb[t] = *(const uint2*)(CB + t * 4);      // 8B codebook entry per thread

    const int bm   = blockIdx.x * TM;
    const int bn   = blockIdx.y * TN;
    const int wave = t >> 5;
    const int lane = t & 31;
    const int wm   = (wave & 1) * 64;         // wave offset in M (2 waves)
    const int wn   = (wave >> 1) * 64;        // wave offset in N (4 waves)
    const int lhalf = lane >> 4;              // 0 -> lanes 0-15, 1 -> lanes 16-31
    const int lrow  = lane & 15;

    v8f acc[4][4];
    #pragma unroll
    for (int fm = 0; fm < 4; ++fm)
        #pragma unroll
        for (int fn = 0; fn < 4; ++fn)
            acc[fm][fn] = (v8f)(0.0f);

    const int NSTEPS = D_IN / TK;             // 128
    if (wave == 0)                            // prologue: DMA first A tile
        tdm_load_a_tile(A + (size_t)bm * D_IN, &Atile[0][0]);

    for (int s = 0; s < NSTEPS; ++s) {
        const int k0 = s * TK;
        __syncthreads();   // prior compute done: Btile and Atile[(s+1)&1] are free

        // --- decode B tile: 256 rows x 8 codes, one row (8 codes) per thread
        {
            const int* qrow = Qidxs + (size_t)(bn + t) * (D_IN / 4) + (k0 >> 2);
            const uint4 q0 = *(const uint4*)(qrow);
            const uint4 q1 = *(const uint4*)(qrow + 4);
            uint2* dst = (uint2*)(Btile + t * LDT);
            dst[0] = cb[q0.x & 255];
            dst[1] = cb[q0.y & 255];
            dst[2] = cb[q0.z & 255];
            dst[3] = cb[q0.w & 255];
            dst[4] = cb[q1.x & 255];
            dst[5] = cb[q1.y & 255];
            dst[6] = cb[q1.z & 255];
            dst[7] = cb[q1.w & 255];
        }
        // --- wave 0: wait for A(s), then kick off DMA of A(s+1) -------------
        if (wave == 0) {
            __builtin_amdgcn_s_wait_tensorcnt((short)0);
            if (s + 1 < NSTEPS)
                tdm_load_a_tile(A + (size_t)bm * D_IN + (k0 + TK),
                                &Atile[(s + 1) & 1][0]);
        }
        __syncthreads();

        const _Float16* At = &Atile[s & 1][0];

        // --- fragment loads (ISA 16-bit A/B VGPR layouts) -------------------
        // A 16x32: lanes 0-15 hold K0-7 then K16-23; lanes 16-31 K8-15 / K24-31
        Frag16 fa[4];
        #pragma unroll
        for (int fm = 0; fm < 4; ++fm) {
            const int m  = wm + fm * 16 + lrow;
            const int ka = lhalf ? 8 : 0;
            fa[fm].q[0] = *(const uint4*)(At + m * LDT + ka);
            fa[fm].q[1] = *(const uint4*)(At + m * LDT + ka + 16);
        }
        // B 32x16: lanes 0-15 hold K0-7 then K16-23 for N=lane; lanes 16-31 K8-15 / K24-31
        Frag16 fb[4];
        #pragma unroll
        for (int fn = 0; fn < 4; ++fn) {
            const int n  = wn + fn * 16 + lrow;
            const int kb = lhalf ? 8 : 0;
            fb[fn].q[0] = *(const uint4*)(Btile + n * LDT + kb);
            fb[fn].q[1] = *(const uint4*)(Btile + n * LDT + kb + 16);
        }
        // --- 16 WMMAs per wave per K-step -----------------------------------
        #pragma unroll
        for (int fm = 0; fm < 4; ++fm)
            #pragma unroll
            for (int fn = 0; fn < 4; ++fn)
                acc[fm][fn] = __builtin_amdgcn_wmma_f32_16x16x32_f16(
                    false, fa[fm].h, false, fb[fn].h,
                    (short)0, acc[fm][fn], false, false);
    }

    // --- epilogue: C/D layout: VGPR r -> M = frag_m + lhalf*8 + r, N = lrow -
    #pragma unroll
    for (int fm = 0; fm < 4; ++fm) {
        #pragma unroll
        for (int fn = 0; fn < 4; ++fn) {
            const int m0 = bm + wm + fm * 16 + lhalf * 8;
            const int n  = bn + wn + fn * 16 + lrow;
            #pragma unroll
            for (int r = 0; r < 8; ++r)
                Z[(size_t)(m0 + r) * M_OUT + n] = (_Float16)acc[fm][fn][r];
        }
    }
}

// ---------------------------------------------------------------------------
// Pass C: per-row  z*(Wscale*1024) -> FWHT(4096) -> *(1/64)*SV -> f32
// ---------------------------------------------------------------------------
__global__ __launch_bounds__(256) void qd4_fwht_out(
    const _Float16* __restrict__ Z, const float* __restrict__ SV,
    const float* __restrict__ Wscale, float* __restrict__ out)
{
    __shared__ float s[M_OUT];
    const int row = blockIdx.x;
    const size_t base = (size_t)row * M_OUT;
    const float ws = Wscale[0] * 1024.0f;
    for (int i = threadIdx.x; i < M_OUT; i += 256)
        s[i] = (float)Z[base + i] * ws;
    __syncthreads();
    for (int h = 1; h < M_OUT; h <<= 1) {
        for (int p = threadIdx.x; p < M_OUT / 2; p += 256) {
            const int i = ((p & ~(h - 1)) << 1) | (p & (h - 1));
            const int j = i + h;
            const float a = s[i], b = s[j];
            s[i] = a + b;
            s[j] = a - b;
        }
        __syncthreads();
    }
    const float sc = 1.0f / 64.0f;
    for (int i = threadIdx.x; i < M_OUT; i += 256)
        out[base + i] = s[i] * sc * SV[i];
}

// ---------------------------------------------------------------------------
extern "C" void kernel_launch(void* const* d_in, const int* in_sizes, int n_in,
                              void* d_out, int out_size, void* d_ws, size_t ws_size,
                              hipStream_t stream) {
    const float*    input  = (const float*)d_in[0];      // (4,2048,4096) f32
    const int*      Qidxs  = (const int*)d_in[1];        // (4096,1024) i32
    const float*    SU     = (const float*)d_in[2];      // (4096,) f32
    const float*    SV     = (const float*)d_in[3];      // (4096,) f32
    const float*    Wscale = (const float*)d_in[4];      // (1,) f32
    const _Float16* CB     = (const _Float16*)d_in[5];   // (256,4) fp16

    float* out = (float*)d_out;                          // (4,2048,4096) f32

    _Float16* Afp16 = (_Float16*)d_ws;                           // 64 MiB
    _Float16* Zfp16 = Afp16 + (size_t)NROWS * D_IN;              // 64 MiB

    qd4_fwht_in <<<NROWS, 256, 0, stream>>>(input, SU, Afp16);
    qd4_gemm    <<<dim3(NROWS / TM, M_OUT / TN), 256, 0, stream>>>(Afp16, Qidxs, CB, Zfp16);
    qd4_fwht_out<<<NROWS, 256, 0, stream>>>(Zfp16, SV, Wscale, out);
}
